// Block_56968446214461
// MI455X (gfx1250) — matrix operand order repaired and checked by reading, not compiled, not run on hardware
//
#include <hip/hip_runtime.h>
#include <math.h>

// ---------------------------------------------------------------------------
// MI455X / gfx1250 implementation of the GCN+Attention block.
// All GEMMs use v_wmma_f32_16x16x32_bf16 (bf16 in, fp32 accumulate).
// qkv/proj are row-flat GEMMs (64 rows/block) to minimize L2 weight traffic;
// the batch-structured attention core is a wave-private WMMA kernel.
// Required d_ws size: ~940 MB.
// ---------------------------------------------------------------------------

typedef __attribute__((ext_vector_type(16))) __bf16 v16bf;
typedef __attribute__((ext_vector_type(8)))  float  v8f;

union F16x16 { v16bf v; unsigned short u[16]; uint4 q[2]; };

__device__ __forceinline__ unsigned short f2bf(float f) {
  unsigned int x = __float_as_uint(f);
  unsigned int r = x + 0x7FFFu + ((x >> 16) & 1u);   // round-to-nearest-even
  return (unsigned short)(r >> 16);
}
__device__ __forceinline__ float bf2f(unsigned short u) {
  return __uint_as_float(((unsigned int)u) << 16);
}
__device__ __forceinline__ float gelu_exact(float x) {
  return 0.5f * x * (1.0f + erff(x * 0.70710678118654752f));
}

__device__ __forceinline__ v8f wmma_bf16(v16bf a, v16bf b, v8f c) {
  return __builtin_amdgcn_wmma_f32_16x16x32_bf16(false, a, false, b, (short)0, c,
                                                 false, false);
}

// A fragment: row-major [M x K] bf16, lane holds row m0+(lane&15).
// Per ISA layout: elems 0..7 at K=k0+8*half, elems 8..15 at K=k0+16+8*half.
__device__ __forceinline__ v16bf load_frag_A(const unsigned short* A, int lda,
                                             int m0, int k0, int lane) {
  const unsigned short* p = A + (size_t)(m0 + (lane & 15)) * lda + k0 + 8 * (lane >> 4);
  F16x16 f;
  f.q[0] = *(const uint4*)p;
  f.q[1] = *(const uint4*)(p + 16);
  return f.v;
}
// B fragment: weight stored [N x K] row-major (= y = x @ W^T layout).
// Lane holds column n0+(lane&15): elems 0..7 at K=k0+8*half, elems 8..15 at K=k0+16+8*half.
__device__ __forceinline__ v16bf load_frag_B(const unsigned short* W, int ldk,
                                             int n0, int k0, int lane) {
  const unsigned short* p = W + (size_t)(n0 + (lane & 15)) * ldk + k0 + 8 * (lane >> 4);
  F16x16 f;
  f.q[0] = *(const uint4*)p;
  f.q[1] = *(const uint4*)(p + 16);
  return f.v;
}

// ---------------------------------------------------------------------------
__global__ void k_cvt(const float* __restrict__ src, unsigned short* __restrict__ dst, int n) {
  int i = blockIdx.x * 256 + threadIdx.x;
  if (i < n) dst[i] = f2bf(src[i]);
}

// ---------------------------------------------------------------------------
// LN1 over J (scale/shift indexed by joint) -> xg bf16 ; LN_A over C -> xa bf16
// grid = B (4096)
// ---------------------------------------------------------------------------
__global__ void k_ln_pre(const float* __restrict__ x,
                         const float* __restrict__ g1, const float* __restrict__ b1,
                         const float* __restrict__ gA, const float* __restrict__ bA,
                         unsigned short* __restrict__ xg, unsigned short* __restrict__ xa) {
  __shared__ float xs[17 * 512];
  size_t base = (size_t)blockIdx.x * 17 * 512;
  for (int i = threadIdx.x; i < 17 * 512; i += 256) xs[i] = x[base + i];
  __syncthreads();
  int lane = threadIdx.x & 31, wave = threadIdx.x >> 5;
  for (int j = wave; j < 17; j += 8) {
    float s = 0.f, s2 = 0.f;
    for (int c = lane; c < 512; c += 32) { float v = xs[j * 512 + c]; s += v; s2 += v * v; }
    for (int o = 16; o > 0; o >>= 1) { s += __shfl_xor(s, o, 32); s2 += __shfl_xor(s2, o, 32); }
    float m = s * (1.f / 512.f);
    float rinv = rsqrtf(s2 * (1.f / 512.f) - m * m + 1e-5f);
    for (int c = lane; c < 512; c += 32)
      xa[base + j * 512 + c] = f2bf((xs[j * 512 + c] - m) * rinv * gA[c] + bA[c]);
  }
  for (int c = threadIdx.x; c < 512; c += 256) {
    float s = 0.f, s2 = 0.f;
#pragma unroll
    for (int j = 0; j < 17; j++) { float v = xs[j * 512 + c]; s += v; s2 += v * v; }
    float m = s * (1.f / 17.f);
    float rinv = rsqrtf(s2 * (1.f / 17.f) - m * m + 1e-5f);
#pragma unroll
    for (int j = 0; j < 17; j++)
      xg[base + j * 512 + c] = f2bf((xs[j * 512 + c] - m) * rinv * g1[j] + b1[j]);
  }
}

// ---------------------------------------------------------------------------
// GCN (gcn1 / gcn2): 4 batches per block (68 rows padded to 80). grid = 1024.
// One B fragment per K-step feeds 5 M-tile WMMAs; adjacency contraction fp32.
// ---------------------------------------------------------------------------
__global__ void __launch_bounds__(256, 1)
k_gcn(const unsigned short* __restrict__ Ain,
      const unsigned short* __restrict__ Wbf,
      const float* __restrict__ bias,
      const float* __restrict__ adj,
      unsigned short* __restrict__ out) {
  __shared__ alignas(16) unsigned short As[80 * 512];          // 80 KB
  __shared__ alignas(16) unsigned short ystage[8][80][3][16];  // 60 KB
  __shared__ float adjs[3 * 17 * 17];
  int tid = threadIdx.x, lane = tid & 31, wave = tid >> 5;
  int half = lane >> 4, ci = lane & 15;
  size_t rowbase = (size_t)blockIdx.x * 68;

  {
    const uint4* s = (const uint4*)(Ain + rowbase * 512);
    uint4* d = (uint4*)As;
    for (int i = tid; i < 4352; i += 256) d[i] = s[i];
    uint4 z = make_uint4(0u, 0u, 0u, 0u);
    for (int i = 4352 + tid; i < 5120; i += 256) d[i] = z;
  }
  for (int i = tid; i < 3 * 17 * 17; i += 256) adjs[i] = adj[i];
  __syncthreads();

  for (int cb = wave; cb < 32; cb += 8) {
    int c0 = cb * 16;
    for (int k3 = 0; k3 < 3; k3++) {
      int n0 = k3 * 512 + c0;
      v8f acc[5] = {};
#pragma unroll 2
      for (int k0 = 0; k0 < 512; k0 += 32) {
        v16bf b = load_frag_B(Wbf, 512, n0, k0, lane);
#pragma unroll
        for (int mt = 0; mt < 5; mt++)
          acc[mt] = wmma_bf16(load_frag_A(As, 512, mt * 16, k0, lane), b, acc[mt]);
      }
      float bia = bias[n0 + ci];
#pragma unroll
      for (int mt = 0; mt < 5; mt++)
#pragma unroll
        for (int r = 0; r < 8; r++) {
          int m = mt * 16 + r + 8 * half;
          ystage[wave][m][k3][ci] = f2bf(acc[mt][r] + bia);
        }
    }
    for (int idx = lane; idx < 4 * 17 * 16; idx += 32) {
      int cc = idx & 15;
      int t = idx >> 4;
      int bb = t / 17, w = t - 17 * bb;
      float s = 0.f;
#pragma unroll
      for (int k3 = 0; k3 < 3; k3++)
#pragma unroll
        for (int j = 0; j < 17; j++)
          s += bf2f(ystage[wave][bb * 17 + j][k3][cc]) * adjs[(k3 * 17 + j) * 17 + w];
      out[(rowbase + bb * 17 + w) * 512 + c0 + cc] = f2bf(s);
    }
  }
}

// ---------------------------------------------------------------------------
// qkv: flat GEMM over 69632 rows, 64 rows/block. grid = 1088.
// Epilogue: +bias, q scaled by D^-0.5, split into q/k/v bf16 buffers.
// ---------------------------------------------------------------------------
__global__ void __launch_bounds__(256, 1)
k_qkv(const unsigned short* __restrict__ xa,
      const unsigned short* __restrict__ Wqkv, const float* __restrict__ bqkv,
      unsigned short* __restrict__ qg, unsigned short* __restrict__ kg,
      unsigned short* __restrict__ vg) {
  __shared__ alignas(16) unsigned short a0[64 * 512];  // 64 KB
  int tid = threadIdx.x, lane = tid & 31, wave = tid >> 5;
  int half = lane >> 4, ci = lane & 15;
  size_t rowbase = (size_t)blockIdx.x * 64;
  {
    const uint4* s = (const uint4*)(xa + rowbase * 512);
    uint4* d = (uint4*)a0;
    for (int i = tid; i < 4096; i += 256) d[i] = s[i];
  }
  __syncthreads();
  for (int t = wave; t < 96; t += 8) {
    int n0 = t * 16;
    v8f acc[4] = {};
#pragma unroll 2
    for (int k0 = 0; k0 < 512; k0 += 32) {
      v16bf b = load_frag_B(Wqkv, 512, n0, k0, lane);
#pragma unroll
      for (int mt = 0; mt < 4; mt++)
        acc[mt] = wmma_bf16(load_frag_A(a0, 512, mt * 16, k0, lane), b, acc[mt]);
    }
    int which = n0 >> 9, hd = n0 & 511;
    float bia = bqkv[n0 + ci];
    unsigned short* dst = (which == 0) ? qg : (which == 1) ? kg : vg;
    float scale = (which == 0) ? 0.125f : 1.0f;   // D^-0.5 folded into q
#pragma unroll
    for (int mt = 0; mt < 4; mt++)
#pragma unroll
      for (int r = 0; r < 8; r++) {
        int m = mt * 16 + r + 8 * half;
        dst[(rowbase + m) * 512 + hd + ci] = f2bf((acc[mt][r] + bia) * scale);
      }
  }
}

// ---------------------------------------------------------------------------
// Attention core: 1 batch per block, head h = wave (all wave-private; no
// block barriers). QK^T and A*V via WMMA on per-wave LDS scratchpads.
// Writes z = xg1 + 0.8*o and o2 = o + 0.5*xg1 (bf16). grid = 4096.
// ---------------------------------------------------------------------------
__global__ void __launch_bounds__(256, 1)
k_attn_core(const unsigned short* __restrict__ qg,
            const unsigned short* __restrict__ kg,
            const unsigned short* __restrict__ vg,
            const unsigned short* __restrict__ xg1,
            unsigned short* __restrict__ z,
            unsigned short* __restrict__ o2) {
  __shared__ alignas(16) unsigned short qs[8][32 * 64];   // 32 KB
  __shared__ alignas(16) unsigned short ks[8][32 * 64];   // 32 KB
  __shared__ alignas(16) unsigned short vTs[8][64 * 32];  // 32 KB
  __shared__ alignas(16) unsigned short attnL[8][32 * 32];// 16 KB
  __shared__ float sc[8][17][18];
  int lane = threadIdx.x & 31, wave = threadIdx.x >> 5;
  int half = lane >> 4, ci = lane & 15;
  int h = wave;
  size_t rowbase = (size_t)blockIdx.x * 17;

  for (int i = lane; i < 2048; i += 32) { qs[wave][i] = 0; ks[wave][i] = 0; vTs[wave][i] = 0; }
  for (int i = lane; i < 1024; i += 32) attnL[wave][i] = 0;
  // gather this head's 17x64 q/k (vectorized) and v transposed
  for (int i = lane; i < 17 * 8; i += 32) {
    int t = i >> 3, dq = i & 7;
    size_t gb = (rowbase + t) * 512 + h * 64;
    uint4 qv = *((const uint4*)(qg + gb) + dq);
    uint4 kv = *((const uint4*)(kg + gb) + dq);
    *((uint4*)(&qs[wave][t * 64]) + dq) = qv;
    *((uint4*)(&ks[wave][t * 64]) + dq) = kv;
    uint4 vv = *((const uint4*)(vg + gb) + dq);
    const unsigned short* ve = (const unsigned short*)&vv;
#pragma unroll
    for (int e = 0; e < 8; e++) vTs[wave][(dq * 8 + e) * 32 + t] = ve[e];
  }
  // ---- scores = q @ k^T (q pre-scaled) ----
  for (int mt = 0; mt < 2; mt++)
    for (int nt = 0; nt < 2; nt++) {
      v8f acc = {};
#pragma unroll
      for (int kk = 0; kk < 64; kk += 32)
        acc = wmma_bf16(load_frag_A(qs[wave], 64, mt * 16, kk, lane),
                        load_frag_B(ks[wave], 64, nt * 16, kk, lane), acc);
#pragma unroll
      for (int r = 0; r < 8; r++) {
        int m = mt * 16 + r + 8 * half, n = nt * 16 + ci;
        if (m < 17 && n < 17) sc[wave][m][n] = acc[r];
      }
    }
  // ---- masked softmax over 17 ----
  if (lane < 17) {
    float mx = -1e30f;
#pragma unroll
    for (int n = 0; n < 17; n++) mx = fmaxf(mx, sc[wave][lane][n]);
    float e[17], sum = 0.f;
#pragma unroll
    for (int n = 0; n < 17; n++) { e[n] = expf(sc[wave][lane][n] - mx); sum += e[n]; }
    float rs = 1.f / sum;
#pragma unroll
    for (int n = 0; n < 17; n++) attnL[wave][lane * 32 + n] = f2bf(e[n] * rs);
#pragma unroll
    for (int n = 17; n < 32; n++) attnL[wave][lane * 32 + n] = 0;
  }
  // ---- o = attn @ v ; epilogue writes z and o2 ----
  for (int mt = 0; mt < 2; mt++)
    for (int nt = 0; nt < 4; nt++) {
      v8f acc = {};
      acc = wmma_bf16(load_frag_A(attnL[wave], 32, mt * 16, 0, lane),
                      load_frag_B(vTs[wave], 32, nt * 16, 0, lane), acc);
#pragma unroll
      for (int r = 0; r < 8; r++) {
        int m = mt * 16 + r + 8 * half;
        if (m < 17) {
          int c = h * 64 + nt * 16 + ci;
          size_t g = (rowbase + m) * 512 + c;
          float ov = acc[r];
          float f1 = bf2f(xg1[g]);
          z[g]  = f2bf(f1 + 0.8f * ov);   // xg1 + attn2gcn*S_A2G
          o2[g] = f2bf(ov + 0.5f * f1);   // o + fh  (S_G2A)
        }
      }
    }
}

// ---------------------------------------------------------------------------
// proj: flat GEMM over 69632 rows, 64 rows/block. grid = 1088.
// ---------------------------------------------------------------------------
__global__ void __launch_bounds__(256, 1)
k_proj(const unsigned short* __restrict__ o2,
       const unsigned short* __restrict__ Wproj, const float* __restrict__ bproj,
       unsigned short* __restrict__ xaout) {
  __shared__ alignas(16) unsigned short a0[64 * 512];  // 64 KB
  int tid = threadIdx.x, lane = tid & 31, wave = tid >> 5;
  int half = lane >> 4, ci = lane & 15;
  size_t rowbase = (size_t)blockIdx.x * 64;
  {
    const uint4* s = (const uint4*)(o2 + rowbase * 512);
    uint4* d = (uint4*)a0;
    for (int i = tid; i < 4096; i += 256) d[i] = s[i];
  }
  __syncthreads();
  for (int t = wave; t < 32; t += 8) {
    int n0 = t * 16;
    v8f acc[4] = {};
#pragma unroll 2
    for (int k0 = 0; k0 < 512; k0 += 32) {
      v16bf b = load_frag_B(Wproj, 512, n0, k0, lane);
#pragma unroll
      for (int mt = 0; mt < 4; mt++)
        acc[mt] = wmma_bf16(load_frag_A(a0, 512, mt * 16, k0, lane), b, acc[mt]);
    }
    float bia = bproj[n0 + ci];
#pragma unroll
    for (int mt = 0; mt < 4; mt++)
#pragma unroll
      for (int r = 0; r < 8; r++) {
        int m = mt * 16 + r + 8 * half;
        xaout[(rowbase + m) * 512 + n0 + ci] = f2bf(acc[mt][r] + bia);
      }
  }
}

// ---------------------------------------------------------------------------
// y = x + xg2 + xa_out (fp32) ; LN2 over C -> bf16. grid = B.
// ---------------------------------------------------------------------------
__global__ void k_res_ln2(const float* __restrict__ x,
                          const unsigned short* __restrict__ xg2,
                          const unsigned short* __restrict__ xao,
                          const float* __restrict__ g2, const float* __restrict__ b2,
                          float* __restrict__ y, unsigned short* __restrict__ ln2y) {
  __shared__ float ys[17 * 512];
  size_t base = (size_t)blockIdx.x * 17 * 512;
  for (int i = threadIdx.x; i < 17 * 512; i += 256) {
    float v = x[base + i] + bf2f(xg2[base + i]) + bf2f(xao[base + i]);
    ys[i] = v;
    y[base + i] = v;
  }
  __syncthreads();
  int lane = threadIdx.x & 31, wave = threadIdx.x >> 5;
  for (int j = wave; j < 17; j += 8) {
    float s = 0.f, s2 = 0.f;
    for (int c = lane; c < 512; c += 32) { float v = ys[j * 512 + c]; s += v; s2 += v * v; }
    for (int o = 16; o > 0; o >>= 1) { s += __shfl_xor(s, o, 32); s2 += __shfl_xor(s2, o, 32); }
    float m = s * (1.f / 512.f);
    float rinv = rsqrtf(s2 * (1.f / 512.f) - m * m + 1e-5f);
    for (int c = lane; c < 512; c += 32)
      ln2y[base + j * 512 + c] = f2bf((ys[j * 512 + c] - m) * rinv * g2[c] + b2[c]);
  }
}

// ---------------------------------------------------------------------------
// MLP: 64 rows/block, fused 512->256 GELU -> 256->256 GELU + res -> 256->512
// GELU; out = y + mlp. grid = 1088.
// ---------------------------------------------------------------------------
__global__ void __launch_bounds__(256, 1)
k_mlp(const unsigned short* __restrict__ a0g,
      const unsigned short* __restrict__ W1, const float* __restrict__ b1,
      const unsigned short* __restrict__ W2, const float* __restrict__ b2,
      const unsigned short* __restrict__ W3, const float* __restrict__ b3,
      const float* __restrict__ yres, float* __restrict__ out) {
  __shared__ alignas(16) unsigned short a0[64 * 512];  // 64 KB
  __shared__ alignas(16) unsigned short h1[64 * 256];  // 32 KB
  __shared__ alignas(16) unsigned short h2[64 * 256];  // 32 KB
  int tid = threadIdx.x, lane = tid & 31, wave = tid >> 5;
  int half = lane >> 4, ci = lane & 15;
  size_t rowbase = (size_t)blockIdx.x * 64;
  {
    const uint4* s = (const uint4*)(a0g + rowbase * 512);
    uint4* d = (uint4*)a0;
    for (int i = tid; i < 4096; i += 256) d[i] = s[i];
  }
  __syncthreads();
  for (int t = wave; t < 16; t += 8) {
    int n0 = t * 16;
    v8f acc[4] = {};
#pragma unroll 2
    for (int k0 = 0; k0 < 512; k0 += 32) {
      v16bf b = load_frag_B(W1, 512, n0, k0, lane);
#pragma unroll
      for (int mt = 0; mt < 4; mt++)
        acc[mt] = wmma_bf16(load_frag_A(a0, 512, mt * 16, k0, lane), b, acc[mt]);
    }
    float bia = b1[n0 + ci];
#pragma unroll
    for (int mt = 0; mt < 4; mt++)
#pragma unroll
      for (int r = 0; r < 8; r++) {
        int m = mt * 16 + r + 8 * half;
        h1[m * 256 + n0 + ci] = f2bf(gelu_exact(acc[mt][r] + bia));
      }
  }
  __syncthreads();
  for (int t = wave; t < 16; t += 8) {
    int n0 = t * 16;
    v8f acc[4] = {};
#pragma unroll 2
    for (int k0 = 0; k0 < 256; k0 += 32) {
      v16bf b = load_frag_B(W2, 256, n0, k0, lane);
#pragma unroll
      for (int mt = 0; mt < 4; mt++)
        acc[mt] = wmma_bf16(load_frag_A(h1, 256, mt * 16, k0, lane), b, acc[mt]);
    }
    float bia = b2[n0 + ci];
#pragma unroll
    for (int mt = 0; mt < 4; mt++)
#pragma unroll
      for (int r = 0; r < 8; r++) {
        int m = mt * 16 + r + 8 * half;
        float v = gelu_exact(acc[mt][r] + bia) + bf2f(h1[m * 256 + n0 + ci]);
        h2[m * 256 + n0 + ci] = f2bf(v);
      }
  }
  __syncthreads();
  for (int t = wave; t < 32; t += 8) {
    int n0 = t * 16;
    v8f acc[4] = {};
#pragma unroll 2
    for (int k0 = 0; k0 < 256; k0 += 32) {
      v16bf b = load_frag_B(W3, 256, n0, k0, lane);
#pragma unroll
      for (int mt = 0; mt < 4; mt++)
        acc[mt] = wmma_bf16(load_frag_A(h2, 256, mt * 16, k0, lane), b, acc[mt]);
    }
    float bia = b3[n0 + ci];
#pragma unroll
    for (int mt = 0; mt < 4; mt++)
#pragma unroll
      for (int r = 0; r < 8; r++) {
        int m = mt * 16 + r + 8 * half;
        size_t g = (rowbase + m) * 512 + n0 + ci;
        out[g] = gelu_exact(acc[mt][r] + bia) + yres[g];
      }
  }
}

// ---------------------------------------------------------------------------
extern "C" void kernel_launch(void* const* d_in, const int* in_sizes, int n_in,
                              void* d_out, int out_size, void* d_ws, size_t ws_size,
                              hipStream_t stream) {
  (void)in_sizes; (void)n_in; (void)out_size; (void)ws_size;
  const float* x      = (const float*)d_in[0];
  const float* adj    = (const float*)d_in[1];
  const float* ln1_g  = (const float*)d_in[2];
  const float* ln1_b  = (const float*)d_in[3];
  const float* lnA_g  = (const float*)d_in[4];
  const float* lnA_b  = (const float*)d_in[5];
  const float* ln2_g  = (const float*)d_in[6];
  const float* ln2_b  = (const float*)d_in[7];
  const float* qkv_w  = (const float*)d_in[8];
  const float* qkv_b  = (const float*)d_in[9];
  const float* proj_w = (const float*)d_in[10];
  const float* proj_b = (const float*)d_in[11];
  const float* gcn1_w = (const float*)d_in[12];
  const float* gcn1_b = (const float*)d_in[13];
  const float* gcn2_w = (const float*)d_in[14];
  const float* gcn2_b = (const float*)d_in[15];
  const float* m1_w   = (const float*)d_in[16];
  const float* m1_b   = (const float*)d_in[17];
  const float* m2_w   = (const float*)d_in[18];
  const float* m2_b   = (const float*)d_in[19];
  const float* m3_w   = (const float*)d_in[20];
  const float* m3_b   = (const float*)d_in[21];

  const size_t NE = (size_t)4096 * 17 * 512;
  char* ws = (char*)d_ws;
  size_t off = 0;
  auto carve = [&](size_t bytes) { void* p = ws + off; off += (bytes + 255) & ~(size_t)255; return p; };

  unsigned short* wqkv = (unsigned short*)carve(1536 * 512 * 2);
  unsigned short* wprj = (unsigned short*)carve(512 * 512 * 2);
  unsigned short* wg1  = (unsigned short*)carve(1536 * 512 * 2);
  unsigned short* wg2  = (unsigned short*)carve(1536 * 512 * 2);
  unsigned short* wm1  = (unsigned short*)carve(256 * 512 * 2);
  unsigned short* wm2  = (unsigned short*)carve(256 * 256 * 2);
  unsigned short* wm3  = (unsigned short*)carve(512 * 256 * 2);
  unsigned short* xg    = (unsigned short*)carve(NE * 2);
  unsigned short* xa    = (unsigned short*)carve(NE * 2);
  unsigned short* xg1   = (unsigned short*)carve(NE * 2);
  unsigned short* zbuf  = (unsigned short*)carve(NE * 2);
  unsigned short* xg2   = (unsigned short*)carve(NE * 2);
  unsigned short* xaout = (unsigned short*)carve(NE * 2);
  unsigned short* ln2y  = (unsigned short*)carve(NE * 2);
  unsigned short* qsb   = (unsigned short*)carve(NE * 2);
  unsigned short* ksb   = (unsigned short*)carve(NE * 2);
  unsigned short* vsb   = (unsigned short*)carve(NE * 2);
  unsigned short* o2b   = (unsigned short*)carve(NE * 2);
  float*          yres  = (float*)carve(NE * 4);

  auto cvt = [&](const float* s, unsigned short* d, int n) {
    k_cvt<<<(n + 255) / 256, 256, 0, stream>>>(s, d, n);
  };
  cvt(qkv_w, wqkv, 1536 * 512);
  cvt(proj_w, wprj, 512 * 512);
  cvt(gcn1_w, wg1, 1536 * 512);
  cvt(gcn2_w, wg2, 1536 * 512);
  cvt(m1_w, wm1, 256 * 512);
  cvt(m2_w, wm2, 256 * 256);
  cvt(m3_w, wm3, 512 * 256);

  k_ln_pre<<<4096, 256, 0, stream>>>(x, ln1_g, ln1_b, lnA_g, lnA_b, xg, xa);
  k_gcn<<<1024, 256, 0, stream>>>(xg, wg1, gcn1_b, adj, xg1);
  k_qkv<<<1088, 256, 0, stream>>>(xa, wqkv, qkv_b, qsb, ksb, vsb);
  k_attn_core<<<4096, 256, 0, stream>>>(qsb, ksb, vsb, xg1, zbuf, o2b);
  k_proj<<<1088, 256, 0, stream>>>(o2b, wprj, proj_b, xaout);
  k_gcn<<<1024, 256, 0, stream>>>(zbuf, wg2, gcn2_b, adj, xg2);
  k_res_ln2<<<4096, 256, 0, stream>>>(x, xg2, xaout, ln2_g, ln2_b, yres, ln2y);
  k_mlp<<<1088, 256, 0, stream>>>(ln2y, wm1, m1_b, wm2, m2_b, wm3, m3_b, yres,
                                  (float*)d_out);
}
